// MixedRepeatHeads_76811195122560
// MI455X (gfx1250) — matrix operand order repaired and checked by reading, not compiled, not run on hardware
//
#include <hip/hip_runtime.h>
#include <cstdint>
#include <cstddef>

// ---------------------------------------------------------------------------
// MixedRepeatHeads for MI455X (gfx1250, wave32).
//   pre-pass: split x / proj_w / out_w into (hi,lo) bf16 pairs (one-time)
//   GEMM1: hidden(hi,lo) = mix-epilogue( x @ proj_w^T )   [split-bf16 3-pass]
//   GEMM2: out = hidden @ out_w^T + out_b
// Inner GEMM loops are pure: async global->LDS (ASYNCcnt, double-buffered)
// + ds_load_b128 fragments + v_wmma_f32_16x16x32_bf16. No conversion VALU
// on the critical path.
// Workspace requirement: ~385 MB (hi/lo copies + hidden).
// ---------------------------------------------------------------------------

#define DIM_        4096
#define HIDDEN_N    256
#define N_HEADS_    16
#define BATCH_      8192
#define SEQ_LEN_    4096
#define DECAY_CONST 8

#define BM   128      // block tile M
#define BN   128      // block tile N
#define BK   32       // K step (bf16 WMMA depth)
#define LDSP 40       // padded LDS row stride in bf16 elems (80B = 20 banks)

typedef __attribute__((ext_vector_type(16))) __bf16 v16bf;
typedef __attribute__((ext_vector_type(8)))  __bf16 v8bf;
typedef __attribute__((ext_vector_type(8)))  float  v8f;

// async 16B global->LDS copy: vdst = per-lane LDS byte offset,
// vaddr = per-lane byte offset, saddr = uniform 64-bit base (SGPR pair)
#define ASYNC_CP16(ldsoff, goff, base)                                       \
  asm volatile("global_load_async_to_lds_b128 %0, %1, %2"                    \
               :: "v"(ldsoff), "v"(goff), "s"(base) : "memory")
#define WAIT_ASYNC(n)                                                        \
  asm volatile("s_wait_asynccnt %0" :: "n"(n) : "memory")

// float -> bf16 (round to nearest even), raw bits
static __device__ __forceinline__ unsigned short f2bf(float f) {
  unsigned u = __builtin_bit_cast(unsigned, f);
  u += 0x7FFFu + ((u >> 16) & 1u);
  return (unsigned short)(u >> 16);
}
static __device__ __forceinline__ float bf2f(unsigned short h) {
  unsigned u = ((unsigned)h) << 16;
  return __builtin_bit_cast(float, u);
}
static __device__ __forceinline__ void split2(float f0, float f1,
                                              unsigned& hi, unsigned& lo) {
  unsigned short h0 = f2bf(f0), h1 = f2bf(f1);
  unsigned short l0 = f2bf(f0 - bf2f(h0)), l1 = f2bf(f1 - bf2f(h1));
  hi = (unsigned)h0 | ((unsigned)h1 << 16);
  lo = (unsigned)l0 | ((unsigned)l1 << 16);
}

#define SHUF16(a, b) __builtin_shufflevector((a), (b), 0,1,2,3,4,5,6,7,8,9,10,11,12,13,14,15)

// ---------------------------------------------------------------------------
// One-time split pre-pass: f32 array -> hi/lo bf16 arrays (packed as uint2)
// ---------------------------------------------------------------------------
__global__ void split_kernel(const float4* __restrict__ src,
                             uint2* __restrict__ dst_hi,
                             uint2* __restrict__ dst_lo, int n4) {
  int i = blockIdx.x * blockDim.x + threadIdx.x;
  const int stride = gridDim.x * blockDim.x;
  for (; i < n4; i += stride) {
    const float4 v = src[i];
    unsigned h0, l0, h1, l1;
    split2(v.x, v.y, h0, l0);
    split2(v.z, v.w, h1, l1);
    uint2 uh; uh.x = h0; uh.y = h1;
    uint2 ul; ul.x = l0; ul.y = l1;
    dst_hi[i] = uh;
    dst_lo[i] = ul;
  }
}

// ---------------------------------------------------------------------------
// Per-head coefficients
// ---------------------------------------------------------------------------
__global__ void coef_kernel(const float* __restrict__ mix_w,
                            const float* __restrict__ mix_b,
                            const float* __restrict__ decay_value,
                            const int*   __restrict__ index,
                            float* __restrict__ coef_w,
                            float* __restrict__ coef_b,
                            float* __restrict__ coef_c) {
  const int h = threadIdx.x;
  if (h >= N_HEADS_) return;
  const int idx = *index;
  const float w = mix_w[(size_t)h * SEQ_LEN_ + idx];
  const float b = mix_b[(size_t)h * SEQ_LEN_ + idx];
  float d = fminf(fmaxf(decay_value[h], 0.9f), 1.0f);
  d = powf(d, 1.0f / (float)DECAY_CONST);
  coef_w[h] = w;
  coef_b[h] = b;
  coef_c[h] = (h < N_HEADS_ / 2) ? (w * d) : d;
}

// ---------------------------------------------------------------------------
// Fused GEMM: C[M,N] = A[M,K] * W[N,K]^T, A/W given as preconverted (hi,lo)
// bf16 arrays. MODE 0: per-head mix epilogue, writes hidden as (hi,lo) bf16.
// MODE 1: +bias epilogue, writes f32 output.
// 256 threads = 8 wave32; 128x128 C-tile; 64x32 per wave (4x2 accumulators);
// async global->LDS double-buffered staging; 3-pass split-bf16 WMMA.
// ---------------------------------------------------------------------------
template <int MODE>
__global__ void __launch_bounds__(256)
fused_gemm_kernel(const unsigned short* __restrict__ Ahi,
                  const unsigned short* __restrict__ Alo,
                  const unsigned short* __restrict__ Bhi,
                  const unsigned short* __restrict__ Blo,
                  const float* __restrict__ proj_b,   // MODE 0
                  const float* __restrict__ cache,    // MODE 0
                  const float* __restrict__ coef_w,   // MODE 0
                  const float* __restrict__ coef_b,   // MODE 0
                  const float* __restrict__ coef_c,   // MODE 0
                  const float* __restrict__ bias,     // MODE 1
                  unsigned short* __restrict__ HidHi, // MODE 0
                  unsigned short* __restrict__ HidLo, // MODE 0
                  float* __restrict__ OutF)           // MODE 1
{
  constexpr int K = DIM_;
  constexpr int NSTEP = K / BK;                 // 128
  constexpr unsigned TILE_BYTES = BM * LDSP * 2; // 10240

  __shared__ unsigned short sAhi[2][BM * LDSP];
  __shared__ unsigned short sAlo[2][BM * LDSP];
  __shared__ unsigned short sBhi[2][BN * LDSP];
  __shared__ unsigned short sBlo[2][BN * LDSP];

  const int tid   = threadIdx.x;
  const int lane  = tid & 31;
  const int wid   = tid >> 5;
  const int wm    = wid >> 2;     // 0..1
  const int wn    = wid & 3;      // 0..3
  const int lrow  = lane & 15;
  const int lhalf = lane >> 4;

  const int mBase = blockIdx.y * BM;
  const int nBase = blockIdx.x * BN;

  // LDS byte offsets (generic-pointer low 32 bits == LDS offset)
  const unsigned ldsAhi = (unsigned)(uintptr_t)(void*)&sAhi[0][0];
  const unsigned ldsAlo = (unsigned)(uintptr_t)(void*)&sAlo[0][0];
  const unsigned ldsBhi = (unsigned)(uintptr_t)(void*)&sBhi[0][0];
  const unsigned ldsBlo = (unsigned)(uintptr_t)(void*)&sBlo[0][0];

  // issue 8 async b128 copies per wave: one 128x32-bf16 tile of each array
  auto stage = [&](int k0, int buf) {
    #pragma unroll
    for (int s = 0; s < 2; ++s) {
      const int i   = tid + s * 256;   // 0..511
      const int row = i >> 2;          // 0..127
      const int seg = i & 3;           // 16B segment within 64B row
      const unsigned gA = (unsigned)(((mBase + row) * K + k0) * 2 + seg * 16);
      const unsigned gB = (unsigned)(((nBase + row) * K + k0) * 2 + seg * 16);
      const unsigned l  = buf * TILE_BYTES + (unsigned)(row * LDSP * 2 + seg * 16);
      ASYNC_CP16(ldsAhi + l, gA, Ahi);
      ASYNC_CP16(ldsAlo + l, gA, Alo);
      ASYNC_CP16(ldsBhi + l, gB, Bhi);
      ASYNC_CP16(ldsBlo + l, gB, Blo);
    }
  };

  v8f acc[4][2] = {};

  stage(0, 0);
  for (int t = 0; t < NSTEP; ++t) {
    const int cur = t & 1;
    if (t + 1 < NSTEP) {
      stage((t + 1) * BK, cur ^ 1);
      WAIT_ASYNC(8);   // tile t's 8 ops done; tile t+1's may be in flight
    } else {
      WAIT_ASYNC(0);
    }
    __syncthreads();

    const unsigned short* pAhi = sAhi[cur];
    const unsigned short* pAlo = sAlo[cur];
    const unsigned short* pBhi = sBhi[cur];
    const unsigned short* pBlo = sBlo[cur];

    // A 16x32 bf16 fragment: lane L<16 -> row L, K=[0..7]+[16..23]; L>=16 -> K=[8..15]+[24..31]
    v16bf ah[4], al[4];
    #pragma unroll
    for (int mi = 0; mi < 4; ++mi) {
      const int base = (wm * 64 + mi * 16 + lrow) * LDSP + lhalf * 8;
      v8bf x0 = *(const v8bf*)&pAhi[base];
      v8bf x1 = *(const v8bf*)&pAhi[base + 16];
      ah[mi] = SHUF16(x0, x1);
      v8bf y0 = *(const v8bf*)&pAlo[base];
      v8bf y1 = *(const v8bf*)&pAlo[base + 16];
      al[mi] = SHUF16(y0, y1);
    }
    // B 32x16 bf16 fragment: lane L -> N=(L&15) (= W row), L<16 -> K=[0..7]+[16..23]; L>=16 -> K=[8..15]+[24..31]
    v16bf bh[2], bl[2];
    #pragma unroll
    for (int ni = 0; ni < 2; ++ni) {
      const int base = (wn * 32 + ni * 16 + lrow) * LDSP + lhalf * 8;
      v8bf x0 = *(const v8bf*)&pBhi[base];
      v8bf x1 = *(const v8bf*)&pBhi[base + 16];
      bh[ni] = SHUF16(x0, x1);
      v8bf y0 = *(const v8bf*)&pBlo[base];
      v8bf y1 = *(const v8bf*)&pBlo[base + 16];
      bl[ni] = SHUF16(y0, y1);
    }

    // 3-pass split accumulation: Ahi*Bhi + Ahi*Blo + Alo*Bhi
    #pragma unroll
    for (int mi = 0; mi < 4; ++mi) {
      #pragma unroll
      for (int ni = 0; ni < 2; ++ni) {
        acc[mi][ni] = __builtin_amdgcn_wmma_f32_16x16x32_bf16(
            false, ah[mi], false, bh[ni], (short)0, acc[mi][ni], false, false);
        acc[mi][ni] = __builtin_amdgcn_wmma_f32_16x16x32_bf16(
            false, ah[mi], false, bl[ni], (short)0, acc[mi][ni], false, false);
        acc[mi][ni] = __builtin_amdgcn_wmma_f32_16x16x32_bf16(
            false, al[mi], false, bh[ni], (short)0, acc[mi][ni], false, false);
      }
    }
    __syncthreads();
  }

  // epilogue; C layout: VGPR j, lane l -> row = 8*(l>>4)+j, col = l&15
  #pragma unroll
  for (int mi = 0; mi < 4; ++mi) {
    const int row0 = mBase + wm * 64 + mi * 16 + lhalf * 8;
    #pragma unroll
    for (int ni = 0; ni < 2; ++ni) {
      const int col = nBase + wn * 32 + ni * 16 + lrow;
      if (MODE == 0) {
        const int h  = col >> 8;
        const int kk = col & (HIDDEN_N - 1);
        const float w  = coef_w[h];
        const float cc = coef_c[h];
        const float bb = coef_b[h];
        const float pb = proj_b[col];
        #pragma unroll
        for (int j = 0; j < 8; ++j) {
          const int row = row0 + j;
          float v = acc[mi][ni][j];
          v = w * (v + pb)
            + cc * cache[((size_t)h * BATCH_ + row) * HIDDEN_N + kk]
            + bb;
          const unsigned short vh = f2bf(v);
          HidHi[(size_t)row * DIM_ + col] = vh;
          HidLo[(size_t)row * DIM_ + col] = f2bf(v - bf2f(vh));
        }
      } else {
        const float bb = bias[col];
        #pragma unroll
        for (int j = 0; j < 8; ++j) {
          const int row = row0 + j;
          OutF[(size_t)row * DIM_ + col] = acc[mi][ni][j] + bb;
        }
      }
    }
  }
}

// ---------------------------------------------------------------------------
extern "C" void kernel_launch(void* const* d_in, const int* in_sizes, int n_in,
                              void* d_out, int out_size, void* d_ws, size_t ws_size,
                              hipStream_t stream) {
  const float* x      = (const float*)d_in[0];
  const float* proj_w = (const float*)d_in[1];  // [16,256,4096] == [4096,4096]
  const float* proj_b = (const float*)d_in[2];
  const float* mix_w  = (const float*)d_in[3];
  const float* mix_b  = (const float*)d_in[4];
  const float* decay  = (const float*)d_in[5];
  const float* cache  = (const float*)d_in[6];
  const float* out_w  = (const float*)d_in[7];
  const float* out_b  = (const float*)d_in[8];
  const int*   index  = (const int*)d_in[9];

  // workspace layout (needs ~385 MB)
  char* ws = (char*)d_ws;
  float* coef_w = (float*)ws;            // 16 floats
  float* coef_b = coef_w + 16;
  float* coef_c = coef_w + 32;
  size_t off = 256;
  const size_t szX = (size_t)BATCH_ * DIM_ * 2;  // 64 MiB
  const size_t szW = (size_t)DIM_   * DIM_ * 2;  // 32 MiB
  unsigned short* x_hi  = (unsigned short*)(ws + off); off += szX;
  unsigned short* x_lo  = (unsigned short*)(ws + off); off += szX;
  unsigned short* w1_hi = (unsigned short*)(ws + off); off += szW;
  unsigned short* w1_lo = (unsigned short*)(ws + off); off += szW;
  unsigned short* w2_hi = (unsigned short*)(ws + off); off += szW;
  unsigned short* w2_lo = (unsigned short*)(ws + off); off += szW;
  unsigned short* hd_hi = (unsigned short*)(ws + off); off += szX;
  unsigned short* hd_lo = (unsigned short*)(ws + off); off += szX;

  // one-time split pre-passes
  const int n4x = BATCH_ * DIM_ / 4;   // 8M float4
  const int n4w = DIM_ * DIM_ / 4;     // 4M float4
  split_kernel<<<4096, 256, 0, stream>>>((const float4*)x,
                                         (uint2*)x_hi, (uint2*)x_lo, n4x);
  split_kernel<<<4096, 256, 0, stream>>>((const float4*)proj_w,
                                         (uint2*)w1_hi, (uint2*)w1_lo, n4w);
  split_kernel<<<4096, 256, 0, stream>>>((const float4*)out_w,
                                         (uint2*)w2_hi, (uint2*)w2_lo, n4w);
  coef_kernel<<<1, 32, 0, stream>>>(mix_w, mix_b, decay, index,
                                    coef_w, coef_b, coef_c);

  dim3 grid(DIM_ / BN, BATCH_ / BM);  // (32, 64)
  fused_gemm_kernel<0><<<grid, 256, 0, stream>>>(
      x_hi, x_lo, w1_hi, w1_lo, proj_b, cache, coef_w, coef_b, coef_c,
      nullptr, hd_hi, hd_lo, nullptr);
  fused_gemm_kernel<1><<<grid, 256, 0, stream>>>(
      hd_hi, hd_lo, w2_hi, w2_lo, nullptr, nullptr, nullptr, nullptr, nullptr,
      out_b, nullptr, nullptr, (float*)d_out);
}
